// PixelChanneledNetwork_67293547593881
// MI455X (gfx1250) — matrix operand order repaired and checked by reading, not compiled, not run on hardware
//
#include <hip/hip_runtime.h>
#include <hip/hip_bf16.h>

typedef __bf16 bf16;
typedef __attribute__((ext_vector_type(16))) __bf16 v16bf;
typedef __attribute__((ext_vector_type(8)))  __bf16 v8bf;
typedef __attribute__((ext_vector_type(4)))  __bf16 v4bf;
typedef __attribute__((ext_vector_type(8)))  float  v8f;
typedef __attribute__((ext_vector_type(4)))  float  v4f;

#define LL   16
#define PP   256      // pixels per output matrix
#define DD   1024     // layer-1 input dim
#define HH   128      // hidden size
#define BB   256      // batch
#define CC   3        // output channels
#define MBLK 128      // batch rows handled per block
#define KPAD 40       // padded K extent of staged weight tile (bf16 halves, 80B row)
#define HPAD 136      // padded row length of h1 (bf16 halves, 272B row)

// ---- fragment assembly helpers (bf16 WMMA VGPR layouts, wave32) ----

__device__ __forceinline__ v16bf make_frag(v8bf lo, v8bf hi) {
  v16bf r;
#pragma unroll
  for (int i = 0; i < 8; ++i) { r[i] = lo[i]; r[i + 8] = hi[i]; }
  return r;
}

// A-matrix 16x32 bf16: per lane, 16 contiguous K halves.
__device__ __forceinline__ v16bf load_afrag(const bf16* p) {
  v8bf lo = *(const v8bf*)(p);
  v8bf hi = *(const v8bf*)(p + 8);
  return make_frag(lo, hi);
}

// B-matrix 32x16 bf16: per lane, 16 contiguous K halves.
__device__ __forceinline__ v16bf load_bfrag(const bf16* p) {
  v8bf lo = *(const v8bf*)(p);
  v8bf hi = *(const v8bf*)(p + 8);
  return make_frag(lo, hi);
}

// Stream one 32(K) x 128(N) fp32 weight tile from global, convert to bf16,
// store transposed [N][K] into LDS. 256 threads: each loads 4x float4
// (coalesced 512B per wave) and emits 4x 8B LDS stores.
__device__ __forceinline__ void stage_tile(const float* __restrict__ src, int kt,
                                           bf16 (*buf)[KPAD], int t) {
  const int n4 = (t & 31) * 4;     // 4 consecutive output columns
  const int k4 = (t >> 5) * 4;     // 4 consecutive K rows
  const float* s = src + (size_t)(kt * 32 + k4) * HH + n4;
  v4bf col[4];
#pragma unroll
  for (int kk = 0; kk < 4; ++kk) {
    v4f row = *(const v4f*)(s + (size_t)kk * HH);
#pragma unroll
    for (int i = 0; i < 4; ++i) col[i][kk] = (bf16)row[i];
  }
#pragma unroll
  for (int i = 0; i < 4; ++i) *(v4bf*)&buf[n4 + i][k4] = col[i];
}

// ---- kernel 1: channel-gather + fp32->bf16 cast of activations ----
__global__ __launch_bounds__(256) void gather_cast_kernel(
    const float* __restrict__ x, bf16* __restrict__ Xc) {
  const int e  = blockIdx.x * blockDim.x + threadIdx.x;  // 0 .. C*B*D/4-1
  const int d4 = e & 255;
  const int b  = (e >> 8) & 255;
  const int c  = e >> 16;
  const int d  = d4 * 4;
  const int j  = d >> 8;
  const int w  = d & 255;
  int m;                                   // idx table {0,1,2,3},{2,3,4,5},{0,1,4,5}
  if (c == 0)      m = j;
  else if (c == 1) m = j + 2;
  else             m = (j < 2) ? j : j + 2;
  const v4f src = *(const v4f*)&x[((size_t)b * 6 + m) * (LL * LL) + w];
  v4bf o;
#pragma unroll
  for (int i = 0; i < 4; ++i) o[i] = (bf16)src[i];
  *(v4bf*)&Xc[((size_t)(c * BB + b)) * DD + d] = o;
}

// ---- kernel 2: per-(c,p) 3-layer MLP via bf16 WMMA ----
__global__ __launch_bounds__(256, 1) void mlp_wmma_kernel(
    const float* __restrict__ W1, const float* __restrict__ B1,
    const float* __restrict__ W2, const float* __restrict__ B2,
    const float* __restrict__ W3, const float* __restrict__ B3,
    const bf16* __restrict__ Xc, float* __restrict__ out) {
  __shared__ bf16 wbuf[2][HH][KPAD];   // double-buffered transposed weight tile
  __shared__ bf16 h1[MBLK][HPAD];      // layer-1 activations (bf16)

  const int mb = blockIdx.x & 1;       // which 128-row batch half
  const int cp = blockIdx.x >> 1;      // (c,p) flat
  const int c  = cp >> 8;
  const int p  = cp & 255;

  const int t    = threadIdx.x;
  const int wv   = t >> 5;             // wave id (0..7) -> 16-row M stripe
  const int lane = t & 31;
  const int lm   = lane & 15;
  const int hf   = lane >> 4;

  const v8f vzero = {0.f, 0.f, 0.f, 0.f, 0.f, 0.f, 0.f, 0.f};
  v8f acc[8];
#pragma unroll
  for (int n = 0; n < 8; ++n) acc[n] = vzero;

  // ---------------- layer 1: [128x1024] x [1024x128] ----------------
  const float* w1p  = W1 + (size_t)cp * (DD * HH);
  const bf16*  arow = Xc + ((size_t)(c * BB + mb * MBLK + wv * 16 + lm)) * DD + hf * 16;

  stage_tile(w1p, 0, wbuf[0], t);
  v16bf af = load_afrag(arow);           // pipelined A fragment (kt = 0)
  __syncthreads();
#pragma unroll 1
  for (int kt = 0; kt < DD / 32; ++kt) {
    const int bsel = kt & 1;
    if (kt + 1 < DD / 32) stage_tile(w1p, kt + 1, wbuf[bsel ^ 1], t);
    if (kt + 2 < DD / 32)                // pull tile kt+2 into L2 (global_prefetch)
      __builtin_prefetch((const char*)(w1p + (size_t)(kt + 2) * 32 * HH) + t * 64, 0, 1);
    v16bf afn = af;
    if (kt + 1 < DD / 32) afn = load_afrag(arow + (kt + 1) * 32);
    // software-pipelined B fragments: load n+1 before issuing WMMA n
    v16bf bcur = load_bfrag(&wbuf[bsel][lm][hf * 16]);
#pragma unroll
    for (int n = 0; n < 8; ++n) {
      v16bf bnxt = bcur;
      if (n < 7) bnxt = load_bfrag(&wbuf[bsel][(n + 1) * 16 + lm][hf * 16]);
      acc[n] = __builtin_amdgcn_wmma_f32_16x16x32_bf16(false, af, false, bcur,
                                                       (short)0, acc[n], false, false);
      bcur = bnxt;
    }
    af = afn;
    __syncthreads();
  }

  // bias + relu -> h1 (bf16, LDS). Accum element (n, r) sits at
  // row = wv*16 + hf*8 + r, col = n*16 + lm.
  const float* b1p = B1 + (size_t)cp * HH;
#pragma unroll
  for (int n = 0; n < 8; ++n) {
    const float bias = b1p[n * 16 + lm];
#pragma unroll
    for (int r = 0; r < 8; ++r) {
      const float v = acc[n][r] + bias;
      h1[wv * 16 + hf * 8 + r][n * 16 + lm] = (bf16)fmaxf(v, 0.f);
    }
    acc[n] = vzero;
  }
  __syncthreads();

  // ---------------- layer 2: [128x128] x [128x128] ----------------
  const float* w2p = W2 + (size_t)cp * (HH * HH);
  stage_tile(w2p, 0, wbuf[0], t);
  __syncthreads();
  af = load_afrag(&h1[wv * 16 + lm][hf * 16]);   // kt = 0 (from LDS, after barrier)
#pragma unroll 1
  for (int kt = 0; kt < HH / 32; ++kt) {
    const int bsel = kt & 1;
    if (kt + 1 < HH / 32) stage_tile(w2p, kt + 1, wbuf[bsel ^ 1], t);
    v16bf afn = af;
    if (kt + 1 < HH / 32) afn = load_afrag(&h1[wv * 16 + lm][(kt + 1) * 32 + hf * 16]);
    v16bf bcur = load_bfrag(&wbuf[bsel][lm][hf * 16]);
#pragma unroll
    for (int n = 0; n < 8; ++n) {
      v16bf bnxt = bcur;
      if (n < 7) bnxt = load_bfrag(&wbuf[bsel][(n + 1) * 16 + lm][hf * 16]);
      acc[n] = __builtin_amdgcn_wmma_f32_16x16x32_bf16(false, af, false, bcur,
                                                       (short)0, acc[n], false, false);
      bcur = bnxt;
    }
    af = afn;
    __syncthreads();
  }

  // ---------------- layer 3: bias2 + relu, dot W3, tanh ----------------
  const float* b2p = B2 + (size_t)cp * HH;
  const float* w3p = W3 + (size_t)cp * HH;
  float s[8];
#pragma unroll
  for (int r = 0; r < 8; ++r) s[r] = 0.f;
#pragma unroll
  for (int n = 0; n < 8; ++n) {
    const float bias = b2p[n * 16 + lm];
    const float w3v  = w3p[n * 16 + lm];   // column of this lane's accum slots
#pragma unroll
    for (int r = 0; r < 8; ++r) {
      const float v = fmaxf(acc[n][r] + bias, 0.f);
      s[r] += v * w3v;
    }
  }
  // reduce over the 16 lanes of each half-wave (columns 0..127 covered)
#pragma unroll
  for (int msk = 1; msk < 16; msk <<= 1) {
#pragma unroll
    for (int r = 0; r < 8; ++r) s[r] += __shfl_xor(s[r], msk, 32);
  }
  if (lm == 0) {
    const float b3v = B3[cp];
    const int row0  = mb * MBLK + wv * 16 + hf * 8;
#pragma unroll
    for (int r = 0; r < 8; ++r) {
      float v = s[r] + b3v;
#if __has_builtin(__builtin_amdgcn_tanhf)
      v = __builtin_amdgcn_tanhf(v);
#else
      v = tanhf(v);
#endif
      out[(size_t)(row0 + r) * (CC * PP) + c * PP + p] = v;
    }
  }
}

extern "C" void kernel_launch(void* const* d_in, const int* in_sizes, int n_in,
                              void* d_out, int out_size, void* d_ws, size_t ws_size,
                              hipStream_t stream) {
  (void)in_sizes; (void)n_in; (void)out_size; (void)ws_size;
  const float* x  = (const float*)d_in[0];
  const float* W1 = (const float*)d_in[1];
  const float* B1 = (const float*)d_in[2];
  const float* W2 = (const float*)d_in[3];
  const float* B2 = (const float*)d_in[4];
  const float* W3 = (const float*)d_in[5];
  const float* B3 = (const float*)d_in[6];
  float* out = (float*)d_out;
  bf16* Xc   = (bf16*)d_ws;   // C*B*D bf16 = 1.5 MB of workspace

  // kernel 1: channel gather + cast (C*B*D/4 threads)
  gather_cast_kernel<<<(CC * BB * DD / 4) / 256, 256, 0, stream>>>(x, Xc);
  // kernel 2: one block per (c, p, batch-half)
  mlp_wmma_kernel<<<CC * PP * 2, 256, 0, stream>>>(W1, B1, W2, B2, W3, B3, Xc, out);
}
